// Cumulative_Probability_Head_81355270521184
// MI455X (gfx1250) — matrix operand order repaired and checked by reading, not compiled, not run on hardware
//
#include <hip/hip_runtime.h>

// ---------------------------------------------------------------------------
// Cumulative probability head for MI455X (gfx1250, wave32, WMMA).
//   out[b, t] = sum_{i<=t} relu(x[b,:]·W_h[i,:] + b_h[i]) + (x[b,:]·W_base + b_base)
// Kernel 1: bf16 WMMA GEMM -> hazards (staged in d_out); blockN==0 blocks also
//           compute base[b] = x[b,:]·W_base in f32 and write it to d_ws.
//           Ping-pong register pipeline + L2 prefetch 2 k-steps ahead.
// Kernel 2: per-row inclusive prefix sum + base add (in-place on d_out).
// ---------------------------------------------------------------------------

typedef __attribute__((ext_vector_type(16))) __bf16 v16bf;
typedef __attribute__((ext_vector_type(8)))  __bf16 v8bf;
typedef __attribute__((ext_vector_type(4)))  __bf16 v4bf;
typedef __attribute__((ext_vector_type(8)))  float  v8f;
typedef __attribute__((ext_vector_type(4)))  float  v4f;

#define B_ROWS  8192
#define DIM_IN  2048
#define DIM_OUT 512

#define BM 128
#define BN 128
#define BK 32
#define LDS_STRIDE 40   // 32 bf16 + 8 pad -> 80B rows (16B aligned, bank-spread)

// ---------------------------------------------------------------------------
// Kernel 1. Grid: (512/128, 8192/128) = (4, 64)  [N fastest -> the 4 blocks
// sharing an x tile are concurrent, so x is fetched from HBM once].
// Block: 256 threads = 8 waves in a 4(M) x 2(N) grid; each wave owns a
// 32x64 tile = 2x4 v_wmma_f32_16x16x32_bf16 accumulators.
// ---------------------------------------------------------------------------
__global__ __launch_bounds__(256)
void hazard_gemm_kernel(const float* __restrict__ x,
                        const float* __restrict__ Wh,
                        const float* __restrict__ bh,
                        const float* __restrict__ Wb,
                        float* __restrict__ out,
                        float* __restrict__ base_ws,   // [8192] or nullptr
                        int fuse_base)
{
    __shared__ __align__(16) __bf16 As[BM * LDS_STRIDE];   // x tile [row][k]
    __shared__ __align__(16) __bf16 Bs[BN * LDS_STRIDE];   // W tile [col][k]
    __shared__ float redbuf[BM * 8];                        // base partials

    const int tid  = threadIdx.x;
    const int lane = tid & 31;
    const int wid  = tid >> 5;
    const int wm   = wid & 3;          // wave M index: 4 waves * 32 rows
    const int wn   = wid >> 2;         // wave N index: 2 waves * 64 cols
    const int l15  = lane & 15;
    const int lhi  = lane >> 4;        // half-wave selector (ISA frag layout)

    const int blockN = blockIdx.x;     // 0..3
    const int blockM = blockIdx.y;     // 0..63

    // Staging map: 8 threads cover one 32-float row segment (128B coalesced);
    // each thread owns one float4 per row, for rows ld_r0 + {0,32,64,96}.
    const int ld_c  = (tid & 7) * 4;   // k column (floats)
    const int ld_r0 = tid >> 3;        // row 0..31

    const float* xg = x  + (size_t)(blockM * BM) * DIM_IN;
    const float* wg = Wh + (size_t)(blockN * BN) * DIM_IN;

    const bool do_base = fuse_base && (blockN == 0);

    v8f acc[2][4];
    #pragma unroll
    for (int mt = 0; mt < 2; ++mt)
        #pragma unroll
        for (int nt = 0; nt < 4; ++nt) {
            v8f z = {};
            acc[mt][nt] = z;
        }
    float basep[4] = {0.f, 0.f, 0.f, 0.f};

    // One pipeline stage: drain (convert+store+base-FMA) the "cur" register
    // set into LDS, prefetch k-step "kn" into the "nxt" set, then run the
    // 8 WMMAs for the tile that was just staged.
    auto stage = [&](v4f (&ca)[4], v4f (&cb)[4], v4f& cwb,
                     v4f (&na)[4], v4f (&nb)[4], v4f& nwb, int kn) {
        // -------- convert current f32 regs -> bf16 -> LDS; base FMAs --------
        #pragma unroll
        for (int i = 0; i < 4; ++i) {
            const int r = ld_r0 + 32 * i;
            v4bf ba, bb;
            ba.x = (__bf16)ca[i].x; ba.y = (__bf16)ca[i].y;
            ba.z = (__bf16)ca[i].z; ba.w = (__bf16)ca[i].w;
            bb.x = (__bf16)cb[i].x; bb.y = (__bf16)cb[i].y;
            bb.z = (__bf16)cb[i].z; bb.w = (__bf16)cb[i].w;
            *(v4bf*)(&As[r * LDS_STRIDE + ld_c]) = ba;   // ds_store_b64
            *(v4bf*)(&Bs[r * LDS_STRIDE + ld_c]) = bb;   // ds_store_b64
            if (do_base) {
                basep[i] += ca[i].x * cwb.x + ca[i].y * cwb.y
                          + ca[i].z * cwb.z + ca[i].w * cwb.w;
            }
        }
        __syncthreads();

        // -------- load next k-step into the idle register set --------
        if (kn < DIM_IN) {
            #pragma unroll
            for (int i = 0; i < 4; ++i) {
                const int r = ld_r0 + 32 * i;
                na[i] = *(const v4f*)(xg + (size_t)r * DIM_IN + kn + ld_c);
                nb[i] = *(const v4f*)(wg + (size_t)r * DIM_IN + kn + ld_c);
            }
            nwb = *(const v4f*)(Wb + kn + ld_c);
        }
        // -------- warm L2 two k-steps ahead (global_prefetch, no LOADcnt) ---
        const int kp = kn + BK;
        if (kp < DIM_IN) {
            #pragma unroll
            for (int i = 0; i < 4; ++i) {
                const int r = ld_r0 + 32 * i;
                __builtin_prefetch(xg + (size_t)r * DIM_IN + kp + ld_c, 0, 1);
                __builtin_prefetch(wg + (size_t)r * DIM_IN + kp + ld_c, 0, 1);
            }
        }

        // -------- build fragments per CDNA5 bf16 WMMA VGPR layouts --------
        // A (16x32): row M = lane&15; K chunks {kb..kb+7, kb+16..kb+23},
        // kb = 8*(lane>=16).
        v16bf afrag[2];
        #pragma unroll
        for (int mt = 0; mt < 2; ++mt) {
            const __bf16* p = &As[(wm * 32 + mt * 16 + l15) * LDS_STRIDE + lhi * 8];
            v8bf lo = *(const v8bf*)(p);        // ds_load_b128
            v8bf hi = *(const v8bf*)(p + 16);   // ds_load_b128
            afrag[mt] = __builtin_shufflevector(lo, hi,
                0,1,2,3,4,5,6,7,8,9,10,11,12,13,14,15);
        }
        // B (32x16): column N = lane&15; K chunks {kb..kb+7, kb+16..kb+23},
        // kb = 8*(lane>=16).
        v16bf bfrag[4];
        #pragma unroll
        for (int nt = 0; nt < 4; ++nt) {
            const __bf16* p = &Bs[(wn * 64 + nt * 16 + l15) * LDS_STRIDE + lhi * 8];
            v8bf lo = *(const v8bf*)(p);        // ds_load_b128
            v8bf hi = *(const v8bf*)(p + 16);   // ds_load_b128
            bfrag[nt] = __builtin_shufflevector(lo, hi,
                0,1,2,3,4,5,6,7,8,9,10,11,12,13,14,15);
        }

        // -------- 8 WMMAs per wave per k-step --------
        #pragma unroll
        for (int mt = 0; mt < 2; ++mt)
            #pragma unroll
            for (int nt = 0; nt < 4; ++nt)
                acc[mt][nt] = __builtin_amdgcn_wmma_f32_16x16x32_bf16(
                    /*neg_a=*/false, afrag[mt],
                    /*neg_b=*/false, bfrag[nt],
                    /*c_mod=*/(short)0, acc[mt][nt],
                    /*reuse_a=*/false, /*reuse_b=*/false);

        __syncthreads();
    };

    // ---- ping-pong register sets; DIM_IN/BK = 64 steps (even) ----
    v4f pa0[4], pb0[4], pwb0;
    v4f pa1[4], pb1[4], pwb1;
    #pragma unroll
    for (int i = 0; i < 4; ++i) {
        const int r = ld_r0 + 32 * i;
        pa0[i] = *(const v4f*)(xg + (size_t)r * DIM_IN + ld_c);
        pb0[i] = *(const v4f*)(wg + (size_t)r * DIM_IN + ld_c);
    }
    pwb0 = *(const v4f*)(Wb + ld_c);

    for (int k0 = 0; k0 < DIM_IN; k0 += 2 * BK) {
        stage(pa0, pb0, pwb0, pa1, pb1, pwb1, k0 + BK);       // even step
        stage(pa1, pb1, pwb1, pa0, pb0, pwb0, k0 + 2 * BK);   // odd step
    }

    // -------- epilogue: bias + ReLU; C layout M = r + 8*lhi, N = lane&15 ----
    #pragma unroll
    for (int mt = 0; mt < 2; ++mt) {
        const int gm_base = blockM * BM + wm * 32 + mt * 16 + lhi * 8;
        #pragma unroll
        for (int nt = 0; nt < 4; ++nt) {
            const int gn = blockN * BN + wn * 64 + nt * 16 + l15;
            const float bias = bh[gn];
            #pragma unroll
            for (int r = 0; r < 8; ++r) {
                float v = acc[mt][nt][r] + bias;
                v = v > 0.0f ? v : 0.0f;
                out[(size_t)(gm_base + r) * DIM_OUT + gn] = v;
            }
        }
    }

    // -------- base GEMV reduction (blockN==0 only; block-uniform branch) ----
    if (do_base) {
        #pragma unroll
        for (int i = 0; i < 4; ++i)
            redbuf[(ld_r0 + 32 * i) * 8 + (tid & 7)] = basep[i];
        __syncthreads();
        if (tid < BM) {
            float s = 0.0f;
            #pragma unroll
            for (int j = 0; j < 8; ++j) s += redbuf[tid * 8 + j];
            base_ws[blockM * BM + tid] = s;
        }
    }
}

// ---------------------------------------------------------------------------
// Kernel 2a: prefix sum only (base comes from d_ws). Grid: 8192 x 256.
// ---------------------------------------------------------------------------
__global__ __launch_bounds__(256)
void cumsum_kernel(const float* __restrict__ base_ws,
                   const float* __restrict__ bb,
                   float* __restrict__ out)
{
    __shared__ float s[256];
    const int b   = blockIdx.x;
    const int tid = threadIdx.x;

    const float base = base_ws[b] + bb[0];

    float* row = out + (size_t)b * DIM_OUT;
    const float2 hv = ((const float2*)row)[tid];
    const float pair = hv.x + hv.y;

    float incl = pair;
    s[tid] = incl;
    __syncthreads();
    #pragma unroll
    for (int off = 1; off < 256; off <<= 1) {
        const float t = (tid >= off) ? s[tid - off] : 0.0f;
        __syncthreads();
        incl += t;
        s[tid] = incl;
        __syncthreads();
    }
    const float excl = incl - pair;

    float2 o;
    o.x = excl + hv.x + base;
    o.y = excl + pair + base;
    ((float2*)row)[tid] = o;
}

// ---------------------------------------------------------------------------
// Kernel 2b: fallback (ws too small): recompute base GEMV from x, then scan.
// ---------------------------------------------------------------------------
__global__ __launch_bounds__(256)
void cumsum_base_kernel(const float* __restrict__ x,
                        const float* __restrict__ Wb,
                        const float* __restrict__ bb,
                        float* __restrict__ out)
{
    __shared__ float s[256];
    const int b   = blockIdx.x;
    const int tid = threadIdx.x;

    const float* xrow = x + (size_t)b * DIM_IN;
    float partial = 0.0f;
    #pragma unroll
    for (int i = 0; i < DIM_IN / 256; ++i)
        partial += xrow[tid + 256 * i] * Wb[tid + 256 * i];

    s[tid] = partial;
    __syncthreads();
    #pragma unroll
    for (int off = 128; off > 0; off >>= 1) {
        if (tid < off) s[tid] += s[tid + off];
        __syncthreads();
    }
    const float base = s[0] + bb[0];
    __syncthreads();

    float* row = out + (size_t)b * DIM_OUT;
    const float2 hv = ((const float2*)row)[tid];
    const float pair = hv.x + hv.y;

    float incl = pair;
    s[tid] = incl;
    __syncthreads();
    #pragma unroll
    for (int off = 1; off < 256; off <<= 1) {
        const float t = (tid >= off) ? s[tid - off] : 0.0f;
        __syncthreads();
        incl += t;
        s[tid] = incl;
        __syncthreads();
    }
    const float excl = incl - pair;

    float2 o;
    o.x = excl + hv.x + base;
    o.y = excl + pair + base;
    ((float2*)row)[tid] = o;
}

// ---------------------------------------------------------------------------
extern "C" void kernel_launch(void* const* d_in, const int* in_sizes, int n_in,
                              void* d_out, int out_size, void* d_ws, size_t ws_size,
                              hipStream_t stream)
{
    const float* x  = (const float*)d_in[0];   // [8192, 2048]
    const float* Wh = (const float*)d_in[1];   // [512, 2048]
    const float* bh = (const float*)d_in[2];   // [512]
    const float* Wb = (const float*)d_in[3];   // [1, 2048]
    const float* bb = (const float*)d_in[4];   // [1]
    float* out = (float*)d_out;                // [8192, 512]

    const int fuse_base = (d_ws != nullptr && ws_size >= (size_t)B_ROWS * sizeof(float));
    float* base_ws = fuse_base ? (float*)d_ws : nullptr;

    dim3 grid(DIM_OUT / BN, B_ROWS / BM);      // N fastest: x-tile sharers run together
    hazard_gemm_kernel<<<grid, 256, 0, stream>>>(x, Wh, bh, Wb, out, base_ws, fuse_base);

    if (fuse_base) {
        cumsum_kernel<<<B_ROWS, 256, 0, stream>>>(base_ws, bb, out);
    } else {
        cumsum_base_kernel<<<B_ROWS, 256, 0, stream>>>(x, Wb, bb, out);
    }
}
